// SimpleGAT_6700148982539
// MI455X (gfx1250) — matrix operand order repaired and checked by reading, not compiled, not run on hardware
//
#include <hip/hip_runtime.h>
#include <stdint.h>

#define IN_DIM   128
#define HID      32
#define HEADS    4
#define F1       (HEADS * HID)   // 128
#define OUT_DIM  64
#define NEG_SLOPE 0.2f

typedef __attribute__((ext_vector_type(16))) _Float16 v16h;
typedef __attribute__((ext_vector_type(8)))  float    v8f;

// ---------------------------------------------------------------------------
// order-preserving float<->uint encoding for atomic max on f32
// ---------------------------------------------------------------------------
__device__ __forceinline__ unsigned enc_f32(float f) {
  unsigned u = __float_as_uint(f);
  return (u & 0x80000000u) ? ~u : (u | 0x80000000u);
}
__device__ __forceinline__ float dec_f32(unsigned u) {
  u = (u & 0x80000000u) ? (u & 0x7FFFFFFFu) : ~u;
  return __uint_as_float(u);
}

// ---------------------------------------------------------------------------
// generic u32 fill (used for zeroing and for -inf max init)
// ---------------------------------------------------------------------------
__global__ void fill_u32_kernel(unsigned* __restrict__ p, unsigned v, long long n) {
  long long i = (long long)blockIdx.x * blockDim.x + threadIdx.x;
  if (i < n) p[i] = v;
}

// ---------------------------------------------------------------------------
// pack W (K x Nout, row-major f32) into WMMA B-fragment order as f16.
// Fragment linear layout: [kt][nt][lane][e], e=0..15 (v16h element).
// B 32x16 f16 layout: N = lane&15, K = kt*32 + (lane<16 ? 0 : 8) + (e<8 ? e : e+8).
// ---------------------------------------------------------------------------
__global__ void pack_b_kernel(const float* __restrict__ W, _Float16* __restrict__ Wfrag,
                              int K, int Nout) {
  int NT = Nout >> 4, KT = K >> 5;
  int t = blockIdx.x * blockDim.x + threadIdx.x;
  int total = KT * NT * 32 * 16;
  if (t >= total) return;
  int e    = t & 15;
  int lane = (t >> 4) & 31;
  int tile = t >> 9;
  int nt   = tile % NT;
  int kt   = tile / NT;
  int k = kt * 32 + ((lane < 16) ? 0 : 8) + ((e < 8) ? e : e + 8);
  int n = nt * 16 + (lane & 15);
  Wfrag[t] = (_Float16)W[(size_t)k * Nout + n];
}

// ---------------------------------------------------------------------------
// C[M x (NT*16)] = A[M x (KT*32)] (f32, converted to f16) @ Bfrag (f16)
// One wave per 16-row M tile; accumulators NT x v8f.
// A 16x32 f16 layout: row = mBase + (lane&15);
//   lane<16 : K = kt*32 + {0..7, 16..23};  lane>=16 : K = kt*32 + {8..15, 24..31}
// ---------------------------------------------------------------------------
template <int KT, int NT>
__global__ void gemm_wmma_f16(const float* __restrict__ A,
                              const _Float16* __restrict__ Bfrag,
                              float* __restrict__ C, int M) {
  const int K    = KT * 32;
  const int Nout = NT * 16;
  int wid  = (int)(((long long)blockIdx.x * blockDim.x + threadIdx.x) >> 5);
  int lane = threadIdx.x & 31;
  int mBase = wid * 16;
  if (mBase >= M) return;

  int row = mBase + (lane & 15);
  if (row >= M) row = M - 1;              // clamp (results masked on store)
  const int kb = (lane < 16) ? 0 : 8;

  v8f acc[NT] = {};

#pragma unroll
  for (int kt = 0; kt < KT; ++kt) {
    const float* ap = A + (size_t)row * K + kt * 32 + kb;
    float4 a0 = *(const float4*)(ap + 0);
    float4 a1 = *(const float4*)(ap + 4);
    float4 a2 = *(const float4*)(ap + 16);
    float4 a3 = *(const float4*)(ap + 20);
    v16h a;
    a[0]  = (_Float16)a0.x; a[1]  = (_Float16)a0.y; a[2]  = (_Float16)a0.z; a[3]  = (_Float16)a0.w;
    a[4]  = (_Float16)a1.x; a[5]  = (_Float16)a1.y; a[6]  = (_Float16)a1.z; a[7]  = (_Float16)a1.w;
    a[8]  = (_Float16)a2.x; a[9]  = (_Float16)a2.y; a[10] = (_Float16)a2.z; a[11] = (_Float16)a2.w;
    a[12] = (_Float16)a3.x; a[13] = (_Float16)a3.y; a[14] = (_Float16)a3.z; a[15] = (_Float16)a3.w;

#pragma unroll
    for (int t = 0; t < NT; ++t) {
      v16h b = *(const v16h*)(Bfrag + (((size_t)kt * NT + t) * 32 + lane) * 16);
      acc[t] = __builtin_amdgcn_wmma_f32_16x16x32_f16(
          /*neg_a=*/false, a, /*neg_b=*/false, b,
          /*c_mod=*/(short)0, acc[t], /*reuse_a=*/false, /*reuse_b=*/false);
    }
  }

  const int col  = lane & 15;
  const int rOff = (lane < 16) ? 0 : 8;
#pragma unroll
  for (int t = 0; t < NT; ++t) {
#pragma unroll
    for (int r = 0; r < 8; ++r) {
      int mr = mBase + rOff + r;
      if (mr < M) C[(size_t)mr * Nout + t * 16 + col] = acc[t][r];
    }
  }
}

// ---------------------------------------------------------------------------
// per-(node, head) attention logits: a_src[n,h] = h[n,h,:] . att_src[h,:]
// ---------------------------------------------------------------------------
__global__ void logits_kernel(const float* __restrict__ hmat,
                              const float* __restrict__ att_s,
                              const float* __restrict__ att_d,
                              float* __restrict__ a_s, float* __restrict__ a_d,
                              int Nn, int H, int C) {
  long long t = (long long)blockIdx.x * blockDim.x + threadIdx.x;
  if (t >= (long long)Nn * H) return;
  int h = (int)(t % H);
  long long n = t / H;
  const float* row = hmat + (size_t)n * H * C + (size_t)h * C;
  float ss = 0.f, sd = 0.f;
  for (int c = 0; c < C; ++c) {
    float v = row[c];
    ss += v * att_s[h * C + c];
    sd += v * att_d[h * C + c];
  }
  a_s[t] = ss;
  a_d[t] = sd;
}

// ---------------------------------------------------------------------------
// pass 1: edge logits + LeakyReLU; store raw, atomic-max into per-dst max.
// edges 0..E-1 from edge_index (int64), edges E..E+N-1 are self loops.
// ---------------------------------------------------------------------------
__global__ void edge_logit_max_kernel(const long long* __restrict__ ei, int E, int N, int H,
                                      const float* __restrict__ a_s,
                                      const float* __restrict__ a_d,
                                      float* __restrict__ e_buf,
                                      unsigned* __restrict__ mEnc) {
  long long t = (long long)blockIdx.x * blockDim.x + threadIdx.x;
  long long total = (long long)(E + N) * H;
  if (t >= total) return;
  int h = (int)(t % H);
  long long e = t / H;
  int src, dst;
  if (e < E) { src = (int)ei[e]; dst = (int)ei[(long long)E + e]; }
  else       { src = dst = (int)(e - E); }
  float v = a_s[(size_t)src * H + h] + a_d[(size_t)dst * H + h];
  v = (v >= 0.f) ? v : NEG_SLOPE * v;
  e_buf[t] = v;
  atomicMax(&mEnc[(size_t)dst * H + h], enc_f32(v));
}

// ---------------------------------------------------------------------------
// pass 2: e = exp(e - max[dst]); atomic-add into denom[dst].
// ---------------------------------------------------------------------------
__global__ void edge_exp_kernel(const long long* __restrict__ ei, int E, int N, int H,
                                const unsigned* __restrict__ mEnc,
                                float* __restrict__ e_buf,
                                float* __restrict__ denom) {
  long long t = (long long)blockIdx.x * blockDim.x + threadIdx.x;
  long long total = (long long)(E + N) * H;
  if (t >= total) return;
  int h = (int)(t % H);
  long long e = t / H;
  int dst = (e < E) ? (int)ei[(long long)E + e] : (int)(e - E);
  float m  = dec_f32(mEnc[(size_t)dst * H + h]);
  float ex = expf(e_buf[t] - m);
  e_buf[t] = ex;
  atomicAdd(&denom[(size_t)dst * H + h], ex);
}

// ---------------------------------------------------------------------------
// pass 3: out[dst] += h[src] * alpha  (one wave32 per edge; lane strides F)
// ---------------------------------------------------------------------------
__global__ void aggregate_kernel(const long long* __restrict__ ei, int E, int N,
                                 int H, int C,
                                 const float* __restrict__ hsrc,
                                 const float* __restrict__ e_buf,
                                 const float* __restrict__ denom,
                                 float* __restrict__ out) {
  long long tid = (long long)blockIdx.x * blockDim.x + threadIdx.x;
  int wid  = (int)(tid >> 5);
  int lane = (int)(tid & 31);
  long long total = (long long)E + N;
  if (wid >= total) return;
  int src, dst;
  if (wid < E) { src = (int)ei[wid]; dst = (int)ei[(long long)E + wid]; }
  else         { src = dst = wid - E; }
  const int F = H * C;
  for (int f = lane; f < F; f += 32) {
    int h = f / C;
    float alpha = e_buf[(size_t)wid * H + h] /
                  (denom[(size_t)dst * H + h] + 1e-16f);
    atomicAdd(&out[(size_t)dst * F + f], hsrc[(size_t)src * F + f] * alpha);
  }
}

// ---------------------------------------------------------------------------
// epilogue: x += bias (broadcast over F); optional ELU
// ---------------------------------------------------------------------------
__global__ void bias_act_kernel(float* __restrict__ x, const float* __restrict__ b,
                                long long n, int F, int do_elu) {
  long long i = (long long)blockIdx.x * blockDim.x + threadIdx.x;
  if (i >= n) return;
  float v = x[i] + b[(int)(i % F)];
  if (do_elu) v = (v > 0.f) ? v : (expf(v) - 1.f);
  x[i] = v;
}

// ---------------------------------------------------------------------------
// host launcher
// ---------------------------------------------------------------------------
static inline int cdivll(long long a, long long b) { return (int)((a + b - 1) / b); }

extern "C" void kernel_launch(void* const* d_in, const int* in_sizes, int n_in,
                              void* d_out, int out_size, void* d_ws, size_t ws_size,
                              hipStream_t stream) {
  const float*     x        = (const float*)d_in[0];
  const long long* ei       = (const long long*)d_in[1];   // int64 [2, E]
  const float*     W1       = (const float*)d_in[2];
  const float*     att_src1 = (const float*)d_in[3];
  const float*     att_dst1 = (const float*)d_in[4];
  const float*     b1       = (const float*)d_in[5];
  const float*     W2       = (const float*)d_in[6];
  const float*     att_src2 = (const float*)d_in[7];
  const float*     att_dst2 = (const float*)d_in[8];
  const float*     b2       = (const float*)d_in[9];
  float* out = (float*)d_out;

  const int N = in_sizes[0] / IN_DIM;
  const int E = in_sizes[1] / 2;
  const long long Etot = (long long)E + N;

  // ---- workspace carve-up (256B aligned), with cross-layer reuse ----
  char* base = (char*)d_ws;
  size_t off = 0;
  auto alloc = [&](size_t bytes) -> char* {
    char* p = base + off;
    off += (bytes + 255) & ~(size_t)255;
    return p;
  };
  _Float16* W1f  = (_Float16*)alloc((size_t)IN_DIM * F1 * sizeof(_Float16));
  _Float16* W2f  = (_Float16*)alloc((size_t)F1 * OUT_DIM * sizeof(_Float16));
  float*    h1   = (float*)alloc((size_t)N * F1 * sizeof(float));   // reused as h2
  float*    aS   = (float*)alloc((size_t)N * HEADS * sizeof(float));
  float*    aD   = (float*)alloc((size_t)N * HEADS * sizeof(float));
  unsigned* mEnc = (unsigned*)alloc((size_t)N * HEADS * sizeof(unsigned));
  float*    den  = (float*)alloc((size_t)N * HEADS * sizeof(float));
  float*    ebuf = (float*)alloc((size_t)Etot * HEADS * sizeof(float));
  float*    agg1 = (float*)alloc((size_t)N * F1 * sizeof(float));
  (void)ws_size;

  const unsigned ENC_NEG_INF = 0x007FFFFFu;  // enc_f32(-inf)
  const int BLK = 256;

  // ---- pack weights into f16 B-fragments ----
  {
    int tot1 = (IN_DIM / 32) * (F1 / 16) * 512;
    pack_b_kernel<<<cdivll(tot1, BLK), BLK, 0, stream>>>(W1, W1f, IN_DIM, F1);
    int tot2 = (F1 / 32) * (OUT_DIM / 16) * 512;
    pack_b_kernel<<<cdivll(tot2, BLK), BLK, 0, stream>>>(W2, W2f, F1, OUT_DIM);
  }

  // ================= layer 1 =================
  {
    long long waves = (N + 15) / 16;
    gemm_wmma_f16<4, 8><<<cdivll(waves * 32, 128), 128, 0, stream>>>(x, W1f, h1, N);
  }
  logits_kernel<<<cdivll((long long)N * HEADS, BLK), BLK, 0, stream>>>(
      h1, att_src1, att_dst1, aS, aD, N, HEADS, HID);

  fill_u32_kernel<<<cdivll((long long)N * HEADS, BLK), BLK, 0, stream>>>(
      mEnc, ENC_NEG_INF, (long long)N * HEADS);
  fill_u32_kernel<<<cdivll((long long)N * HEADS, BLK), BLK, 0, stream>>>(
      (unsigned*)den, 0u, (long long)N * HEADS);
  fill_u32_kernel<<<cdivll((long long)N * F1, BLK), BLK, 0, stream>>>(
      (unsigned*)agg1, 0u, (long long)N * F1);

  edge_logit_max_kernel<<<cdivll(Etot * HEADS, BLK), BLK, 0, stream>>>(
      ei, E, N, HEADS, aS, aD, ebuf, mEnc);
  edge_exp_kernel<<<cdivll(Etot * HEADS, BLK), BLK, 0, stream>>>(
      ei, E, N, HEADS, mEnc, ebuf, den);
  aggregate_kernel<<<cdivll(Etot * 32, BLK), BLK, 0, stream>>>(
      ei, E, N, HEADS, HID, h1, ebuf, den, agg1);

  // += b1, ELU (becomes layer-2 input)
  bias_act_kernel<<<cdivll((long long)N * F1, BLK), BLK, 0, stream>>>(
      agg1, b1, (long long)N * F1, F1, 1);

  // ================= layer 2 =================
  float* h2 = h1;  // reuse
  {
    long long waves = (N + 15) / 16;
    gemm_wmma_f16<4, 4><<<cdivll(waves * 32, 128), 128, 0, stream>>>(agg1, W2f, h2, N);
  }
  logits_kernel<<<cdivll((long long)N, BLK), BLK, 0, stream>>>(
      h2, att_src2, att_dst2, aS, aD, N, 1, OUT_DIM);

  fill_u32_kernel<<<cdivll((long long)N, BLK), BLK, 0, stream>>>(
      mEnc, ENC_NEG_INF, (long long)N);
  fill_u32_kernel<<<cdivll((long long)N, BLK), BLK, 0, stream>>>(
      (unsigned*)den, 0u, (long long)N);
  fill_u32_kernel<<<cdivll((long long)N * OUT_DIM, BLK), BLK, 0, stream>>>(
      (unsigned*)out, 0u, (long long)N * OUT_DIM);

  edge_logit_max_kernel<<<cdivll(Etot, BLK), BLK, 0, stream>>>(
      ei, E, N, 1, aS, aD, ebuf, mEnc);
  edge_exp_kernel<<<cdivll(Etot, BLK), BLK, 0, stream>>>(
      ei, E, N, 1, mEnc, ebuf, den);
  aggregate_kernel<<<cdivll(Etot * 32, BLK), BLK, 0, stream>>>(
      ei, E, N, 1, OUT_DIM, h2, ebuf, den, out);

  // += b2 (mean over 1 head == identity), no activation
  bias_act_kernel<<<cdivll((long long)N * OUT_DIM, BLK), BLK, 0, stream>>>(
      out, b2, (long long)N * OUT_DIM, OUT_DIM, 0);
}
